// Memory_61598420959287
// MI455X (gfx1250) — matrix operand order repaired and checked by reading, not compiled, not run on hardware
//
#include <hip/hip_runtime.h>
#include <hip/hip_bf16.h>

typedef __bf16 bf16;
typedef __bf16 v16bf __attribute__((ext_vector_type(16)));
typedef __bf16 v8bf  __attribute__((ext_vector_type(8)));
typedef __bf16 v4bf  __attribute__((ext_vector_type(4)));
typedef float  v8f   __attribute__((ext_vector_type(8)));

#define B_    4
#define T_    2048
#define C_    1024
#define H_    16
#define DH_   64
#define M_    (B_ * T_)     /* 8192 tokens  */
#define NQKV_ (3 * C_)      /* 3072         */
#define MEMLEN_ 512

__device__ __forceinline__ v16bf cat8(v8bf lo, v8bf hi) {
    return __builtin_shufflevector(lo, hi, 0,1,2,3,4,5,6,7,8,9,10,11,12,13,14,15);
}
__device__ __forceinline__ v8f wmma_bf16(v16bf a, v16bf b, v8f c) {
    // D(f32 16x16) = A(bf16 16x32) * B(bf16 32x16) + C
    return __builtin_amdgcn_wmma_f32_16x16x32_bf16(false, a, false, b, (short)0, c, false, false);
}

// ---------------------------------------------------------------------------
// Prep kernels
// ---------------------------------------------------------------------------
// f32 -> bf16, 4 elements per thread (128-bit loads, 64-bit stores)
__global__ void cvt_bf16_kernel(const float* __restrict__ in, bf16* __restrict__ out, int n4) {
    int i = blockIdx.x * 256 + threadIdx.x;
    if (i < n4) {
        float4 f = ((const float4*)in)[i];
        v4bf o = {(bf16)f.x, (bf16)f.y, (bf16)f.z, (bf16)f.w};
        ((v4bf*)out)[i] = o;
    }
}

// Tiled transpose+convert: Wt[n*1024 + k] = bf16(W[k*N + n]), K fixed at 1024.
// 32x32 f32 tile in LDS, both global sides coalesced.
__global__ __launch_bounds__(256) void wtrans_kernel(
    const float* __restrict__ W, bf16* __restrict__ Wt, int N)
{
    __shared__ float tile[32][33];
    const int nt = blockIdx.x * 32;
    const int kt = blockIdx.y * 32;
    const int tx = threadIdx.x & 31;
    const int ty = threadIdx.x >> 5;      // 0..7
    #pragma unroll
    for (int i = 0; i < 4; ++i)
        tile[ty + i * 8][tx] = W[(size_t)(kt + ty + i * 8) * N + nt + tx];
    __syncthreads();
    #pragma unroll
    for (int i = 0; i < 4; ++i)
        Wt[(size_t)(nt + ty + i * 8) * 1024 + kt + tx] = (bf16)tile[tx][ty + i * 8];
}

// new_memory = x[:, T-512:, :], float4 copy
__global__ void tailcopy_kernel(const float* __restrict__ x, float* __restrict__ o) {
    int i = blockIdx.x * 256 + threadIdx.x;     // < 4*512*1024/4 = 524288
    int b = i >> 17;                            // 131072 float4 per batch
    int r = i & 131071;
    const float4* src = (const float4*)(x + (size_t)b * (T_ * C_) + (size_t)(T_ - MEMLEN_) * C_);
    ((float4*)o)[i] = src[r];
}

// ---------------------------------------------------------------------------
// bf16 WMMA GEMM: out(M x N) = A(M x K) * Bt(N x K)^T + bias
// MODE 0: scatter q,k head-major [B,H,T,Dh]; v TRANSPOSED head-major [B,H,Dh,T]
// MODE 1: f32 output (projection)
// Block: 256 threads = 8 waves; 128x128 tile; K staged 32 at a time in LDS.
// ---------------------------------------------------------------------------
template <int MODE>
__global__ __launch_bounds__(256) void gemm_bf16(
    const bf16* __restrict__ A, const bf16* __restrict__ Bt,
    const float* __restrict__ bias,
    float* __restrict__ outF,
    bf16* __restrict__ qO, bf16* __restrict__ kO, bf16* __restrict__ vO,
    int N, int K)
{
    __shared__ bf16 As[128 * 40];   // padded stride 40 (80B): bank-conflict-free
    __shared__ bf16 Bs[128 * 40];

    const int tid  = threadIdx.x;
    const int lane = tid & 31;
    const int l15  = lane & 15;
    const int half = lane >> 4;
    const int w    = tid >> 5;
    const int wm   = (w >> 2) * 64;     // wave M offset within tile (0,64)
    const int wn   = (w & 3) * 32;      // wave N offset within tile (0..96)
    const int m0   = blockIdx.y * 128;
    const int n0   = blockIdx.x * 128;
    const int row  = tid >> 1;          // 0..127 (staging row)
    const int colh = (tid & 1) * 16;    // 0/16   (staging half-row)

    v8f zero = {};
    v8f acc[4][2];
    #pragma unroll
    for (int i = 0; i < 4; ++i)
        #pragma unroll
        for (int j = 0; j < 2; ++j) acc[i][j] = zero;

    const bf16* ga = A  + (size_t)(m0 + row) * K + colh;
    const bf16* gb = Bt + (size_t)(n0 + row) * K + colh;

    for (int k0 = 0; k0 < K; k0 += 32) {
        uint4 a0 = *(const uint4*)(ga + k0);
        uint4 a1 = *(const uint4*)(ga + k0 + 8);
        uint4 b0 = *(const uint4*)(gb + k0);
        uint4 b1 = *(const uint4*)(gb + k0 + 8);
        if (k0 + 32 < K) {                      // emits global_prefetch_b8
            __builtin_prefetch(ga + k0 + 32, 0, 1);
            __builtin_prefetch(gb + k0 + 32, 0, 1);
        }
        __syncthreads();
        *(uint4*)&As[row * 40 + colh]     = a0;
        *(uint4*)&As[row * 40 + colh + 8] = a1;
        *(uint4*)&Bs[row * 40 + colh]     = b0;
        *(uint4*)&Bs[row * 40 + colh + 8] = b1;
        __syncthreads();

        // A fragment: lanes 0-15 -> K 0..15, lanes 16-31 -> K 16..31
        v16bf af[4];
        #pragma unroll
        for (int mt = 0; mt < 4; ++mt) {
            const bf16* p = &As[(wm + mt * 16 + l15) * 40];
            v8bf lo = *(const v8bf*)(p + half * 16);
            v8bf hi = *(const v8bf*)(p + half * 16 + 8);
            af[mt] = cat8(lo, hi);
        }
        // B fragment: lanes 0-15 -> K 0..15, lanes 16-31 -> K 16..31 (N = lane&15)
        v16bf bfr[2];
        #pragma unroll
        for (int nt = 0; nt < 2; ++nt) {
            const bf16* p = &Bs[(wn + nt * 16 + l15) * 40] + half * 16;
            bfr[nt] = cat8(*(const v8bf*)p, *(const v8bf*)(p + 8));
        }
        #pragma unroll
        for (int mt = 0; mt < 4; ++mt)
            #pragma unroll
            for (int nt = 0; nt < 2; ++nt)
                acc[mt][nt] = wmma_bf16(af[mt], bfr[nt], acc[mt][nt]);
    }

    // Epilogue. C layout: vgpr r, lane l -> row = r + (l>=16)*8, col = l&15
    #pragma unroll
    for (int mt = 0; mt < 4; ++mt) {
        #pragma unroll
        for (int nt = 0; nt < 2; ++nt) {
            const int gn  = n0 + wn + nt * 16 + l15;
            const float bia = bias[gn];
            #pragma unroll
            for (int r = 0; r < 8; ++r) {
                const int gm = m0 + wm + mt * 16 + r + half * 8;
                const float val = acc[mt][nt][r] + bia;
                if (MODE == 1) {
                    outF[(size_t)gm * N + gn] = val;
                } else {
                    const int which = gn >> 10;      // 0=q 1=k 2=v
                    const int c = gn & 1023;
                    const int h = c >> 6;
                    const int d = c & 63;
                    const int b = gm >> 11;          // token / 2048
                    const int t = gm & 2047;
                    if (which == 2) {
                        // V stored transposed: [B,H,Dh,T] for contiguous P*V loads
                        vO[((size_t)(b * H_ + h) * DH_ + d) * T_ + t] = (bf16)val;
                    } else {
                        bf16* dst = (which == 0) ? qO : kO;
                        dst[((size_t)(b * H_ + h) * T_ + t) * DH_ + d] = (bf16)val;
                    }
                }
            }
        }
    }
}

// ---------------------------------------------------------------------------
// Flash attention: one wave (32 threads) per 16-query tile per (b,h).
// Online softmax in f32; scores & P*V via bf16 WMMA; keys 32 per step.
// K is [B,H,T,Dh]; V is transposed [B,H,Dh,T] so all fragments load as
// contiguous 16B chunks (global_load_b128).
// ---------------------------------------------------------------------------
__global__ __launch_bounds__(32) void attn_kernel(
    const bf16* __restrict__ q, const bf16* __restrict__ k, const bf16* __restrict__ vt,
    bf16* __restrict__ y)
{
    __shared__ bf16 P[16 * 40];     // P repack tile (C-layout -> A-layout)

    const int lane = threadIdx.x & 31;
    const int l15  = lane & 15;
    const int half = lane >> 4;
    const int qb   = blockIdx.x;    // query tile (16 rows)
    const int bh   = blockIdx.y;    // b*H + h

    const bf16* qbase  = q  + ((size_t)bh * T_ + qb * 16) * DH_;
    const bf16* kbase  = k  + (size_t)bh * T_ * DH_;
    const bf16* vtbase = vt + (size_t)bh * DH_ * T_;

    // Q fragments for both K=32 chunks of Dh=64 (loaded once, reused)
    v16bf aq[2];
    #pragma unroll
    for (int c = 0; c < 2; ++c) {
        const bf16* p = qbase + l15 * DH_ + c * 32;
        aq[c] = cat8(*(const v8bf*)(p + half * 16), *(const v8bf*)(p + half * 16 + 8));
    }

    v8f zero = {};
    v8f acc[4];
    float mI[8], lI[8];
    #pragma unroll
    for (int r = 0; r < 8; ++r) { mI[r] = -3.0e38f; lI[r] = 0.0f; }
    #pragma unroll
    for (int dt = 0; dt < 4; ++dt) acc[dt] = zero;

    const int kend = qb * 16 + 16;              // exclusive key bound (causal)
    for (int kb = 0; kb < kend; kb += 32) {
        // S0: keys kb..kb+15, S1: keys kb+16..kb+31  (max key 16*qb+31 <= 2047)
        v8f s0 = zero, s1 = zero;
        #pragma unroll
        for (int c = 0; c < 2; ++c) {
            const bf16* p0 = kbase + (size_t)(kb + l15) * DH_ + c * 32 + half * 16;
            s0 = wmma_bf16(aq[c], cat8(*(const v8bf*)p0, *(const v8bf*)(p0 + 8)), s0);
        }
        #pragma unroll
        for (int c = 0; c < 2; ++c) {
            const bf16* p1 = kbase + (size_t)(kb + 16 + l15) * DH_ + c * 32 + half * 16;
            s1 = wmma_bf16(aq[c], cat8(*(const v8bf*)p1, *(const v8bf*)(p1 + 8)), s1);
        }

        // scale, causal mask, online softmax (rows live in 16-lane halves)
        float p0v[8], p1v[8], fr[8];
        #pragma unroll
        for (int r = 0; r < 8; ++r) {
            const int qrow = qb * 16 + r + half * 8;
            float a  = s0[r] * 0.125f;          // 1/sqrt(64)
            float b2 = s1[r] * 0.125f;
            if (kb + l15 > qrow)      a  = -3.0e38f;
            if (kb + 16 + l15 > qrow) b2 = -3.0e38f;
            float mx = fmaxf(a, b2);
            #pragma unroll
            for (int off = 8; off >= 1; off >>= 1)
                mx = fmaxf(mx, __shfl_xor(mx, off, 32));
            const float mn = fmaxf(mI[r], mx);
            const float f  = __expf(mI[r] - mn);
            const float e0 = __expf(a - mn);    // masked -> underflow to 0
            const float e1 = __expf(b2 - mn);
            float rs = e0 + e1;
            #pragma unroll
            for (int off = 8; off >= 1; off >>= 1)
                rs += __shfl_xor(rs, off, 32);
            lI[r] = lI[r] * f + rs;
            mI[r] = mn;
            p0v[r] = e0; p1v[r] = e1; fr[r] = f;
        }
        #pragma unroll
        for (int dt = 0; dt < 4; ++dt)
            #pragma unroll
            for (int r = 0; r < 8; ++r)
                acc[dt][r] *= fr[r];

        // repack P (C-layout) -> LDS -> A-layout fragment
        #pragma unroll
        for (int r = 0; r < 8; ++r) {
            const int rr = r + half * 8;
            P[rr * 40 + l15]      = (bf16)p0v[r];
            P[rr * 40 + 16 + l15] = (bf16)p1v[r];
        }
        asm volatile("s_wait_dscnt 0" ::: "memory");   // wave-local LDS RAW
        v16bf pf = cat8(*(const v8bf*)&P[l15 * 40 + half * 16],
                        *(const v8bf*)&P[l15 * 40 + half * 16 + 8]);

        // V^T block: lane holds dim d = dt*16+l15, keys contiguous -> b128 loads
        #pragma unroll
        for (int dt = 0; dt < 4; ++dt) {
            const bf16* p = vtbase + (size_t)(dt * 16 + l15) * T_ + kb + half * 16;
            v16bf vf = cat8(*(const v8bf*)p, *(const v8bf*)(p + 8));
            acc[dt] = wmma_bf16(pf, vf, acc[dt]);
        }
    }

    // normalize and store y in (B,T,C) bf16 layout (heads re-interleaved)
    const int b = bh >> 4;
    const int h = bh & 15;
    #pragma unroll
    for (int r = 0; r < 8; ++r) {
        const float inv = 1.0f / lI[r];
        const int t = qb * 16 + r + half * 8;
        const size_t base = ((size_t)(b * T_ + t)) * C_ + h * DH_;
        #pragma unroll
        for (int dt = 0; dt < 4; ++dt)
            y[base + dt * 16 + l15] = (bf16)(acc[dt][r] * inv);
    }
}

// ---------------------------------------------------------------------------
extern "C" void kernel_launch(void* const* d_in, const int* in_sizes, int n_in,
                              void* d_out, int out_size, void* d_ws, size_t ws_size,
                              hipStream_t stream)
{
    const float* x      = (const float*)d_in[0];
    const float* W_attn = (const float*)d_in[1];
    const float* b_attn = (const float*)d_in[2];
    const float* W_proj = (const float*)d_in[3];
    const float* b_proj = (const float*)d_in[4];

    float* out    = (float*)d_out;
    float* outMem = out + (size_t)M_ * C_;      // new_memory follows out

    char* ws = (char*)d_ws;
    size_t off = 0;
    auto take = [&](size_t bytes) -> void* {
        void* p = ws + off;
        off += (bytes + 255) & ~(size_t)255;
        return p;
    };
    bf16* xb  = (bf16*)take((size_t)M_ * C_ * 2);       // x in bf16
    bf16* WaT = (bf16*)take((size_t)NQKV_ * C_ * 2);    // W_attn^T bf16 (N x K)
    bf16* WpT = (bf16*)take((size_t)C_ * C_ * 2);       // W_proj^T bf16
    bf16* qB  = (bf16*)take((size_t)M_ * C_ * 2);       // [B,H,T,Dh]
    bf16* kB  = (bf16*)take((size_t)M_ * C_ * 2);       // [B,H,T,Dh]
    bf16* vT  = (bf16*)take((size_t)M_ * C_ * 2);       // [B,H,Dh,T] (transposed)
    bf16* yB  = (bf16*)take((size_t)M_ * C_ * 2);       // attention output, (B*T, C)

    cvt_bf16_kernel<<<(M_ * C_ / 4 + 255) / 256, 256, 0, stream>>>(x, xb, M_ * C_ / 4);
    tailcopy_kernel<<<(B_ * MEMLEN_ * C_ / 4 + 255) / 256, 256, 0, stream>>>(x, outMem);
    wtrans_kernel<<<dim3(NQKV_ / 32, 32), 256, 0, stream>>>(W_attn, WaT, NQKV_);
    wtrans_kernel<<<dim3(C_ / 32, 32), 256, 0, stream>>>(W_proj, WpT, C_);

    gemm_bf16<0><<<dim3(NQKV_ / 128, M_ / 128), 256, 0, stream>>>(
        xb, WaT, b_attn, nullptr, qB, kB, vT, NQKV_, C_);

    attn_kernel<<<dim3(T_ / 16, B_ * H_), 32, 0, stream>>>(qB, kB, vT, yB);

    gemm_bf16<1><<<dim3(C_ / 128, M_ / 128), 256, 0, stream>>>(
        yB, WpT, b_proj, out, nullptr, nullptr, nullptr, C_, C_);
}
